// AttentionDecoder_87978110091937
// MI455X (gfx1250) — matrix operand order repaired and checked by reading, not compiled, not run on hardware
//
#include <hip/hip_runtime.h>
#include <stdint.h>
#include <stddef.h>

// ---------------------------------------------------------------------------
// Types for CDNA5 (gfx1250) WMMA: v_wmma_f32_16x16x32_bf16, wave32.
// ---------------------------------------------------------------------------
typedef unsigned short u16;
typedef unsigned int   u32;
typedef __bf16 bf16x16 __attribute__((ext_vector_type(16)));
typedef float  f32x8   __attribute__((ext_vector_type(8)));
typedef u32    u32x4   __attribute__((ext_vector_type(4)));

union Frag { u32x4 q[2]; bf16x16 v; };

__device__ __forceinline__ u16 f2bf(float f) {
  u32 u = __float_as_uint(f);
  u += 0x7FFFu + ((u >> 16) & 1u);   // round-to-nearest-even
  return (u16)(u >> 16);
}
__device__ __forceinline__ float bf2f(u16 h) { return __uint_as_float((u32)h << 16); }
__device__ __forceinline__ float sigmoidf_(float x) { return 1.0f / (1.0f + __expf(-x)); }

// Problem constants
static constexpr int  Bc = 32, Sc = 256, Tc = 400, Mc = 80, Dc = 512, Hc = 8;
static constexpr long BT = (long)Bc * Tc;   // 12800
static constexpr long BS = (long)Bc * Sc;   // 8192

// ---------------------------------------------------------------------------
// Generic bf16 WMMA GEMM:  C[M,N] = alpha * A[M,K] * W[N,K]^T (+ bias[N])
// One 16x16 C tile per wave; used for batched/odd-shaped GEMMs.
// Fragment layouts per CDNA5 ISA (see comments in round-1 source):
//  A frag lane l: two contiguous 16B runs of row (l&15) at k = (l>>4)*8 (+16)
//  B frag lane l: two contiguous 16B runs of W row (l&15) at k = (l>>4)*8 (+16)
//  C frag VGPR v: row = v + (l>>4)*8, col = l&15
// ---------------------------------------------------------------------------
__global__ __launch_bounds__(256)
void k_gemm_wmma(const u16* __restrict__ A, int lda, long sA0, long sA1,
                 const u16* __restrict__ W, int ldw, long sW0, long sW1,
                 void* __restrict__ Cv, int ldc, long sC0, long sC1,
                 const float* __restrict__ bias,
                 int Mtiles, int Ntiles, int Ksteps, int zdiv,
                 float alpha, int outBf16)
{
  const int wid  = threadIdx.x >> 5;
  const int lane = threadIdx.x & 31;
  long tile = (long)blockIdx.x * 8 + wid;
  long tot  = (long)Mtiles * Ntiles;
  if (tile >= tot) return;
  const int mt = (int)(tile / Ntiles);
  const int nt = (int)(tile % Ntiles);
  const int z  = blockIdx.y;
  const long z0 = z / zdiv, z1 = z % zdiv;
  A += z0 * sA0 + z1 * sA1;
  W += z0 * sW0 + z1 * sW1;

  const int row = lane & 15, hi = lane >> 4;
  const u16* Ap = A + (long)(mt * 16 + row) * lda + hi * 8;
  const u16* Wp = W + (long)(nt * 16 + row) * ldw + hi * 8;

  f32x8 acc = {0.f, 0.f, 0.f, 0.f, 0.f, 0.f, 0.f, 0.f};
  for (int kk = 0; kk < Ksteps; ++kk) {
    Frag a, b;
    a.q[0] = *(const u32x4*)(Ap);
    a.q[1] = *(const u32x4*)(Ap + 16);
    b.q[0] = *(const u32x4*)(Wp);
    b.q[1] = *(const u32x4*)(Wp + 16);
    __builtin_prefetch(Ap + 64, 0, 1);   // global_prefetch_b8
    __builtin_prefetch(Wp + 64, 0, 1);
    Ap += 32; Wp += 32;
    acc = __builtin_amdgcn_wmma_f32_16x16x32_bf16(false, a.v, false, b.v,
                                                  (short)0, acc, false, false);
  }

  const int ccol = nt * 16 + row;
  const float bv = bias ? bias[ccol] : 0.f;
  if (outBf16) {
    u16* C = (u16*)Cv + z0 * sC0 + z1 * sC1;
#pragma unroll
    for (int v = 0; v < 8; ++v) {
      int r = mt * 16 + v + hi * 8;
      C[(long)r * ldc + ccol] = f2bf(acc[v] * alpha + bv);
    }
  } else {
    float* C = (float*)Cv + z0 * sC0 + z1 * sC1;
#pragma unroll
    for (int v = 0; v < 8; ++v) {
      int r = mt * 16 + v + hi * 8;
      C[(long)r * ldc + ccol] = acc[v] * alpha + bv;
    }
  }
}

// ---------------------------------------------------------------------------
// Register-blocked bf16 WMMA GEMM: 64x32 C tile per wave.
// 4 A-fragments x 2 B-fragments -> 8 accumulators, 8 v_wmma per K-step.
// A fragments reused 2x, B fragments reused 4x -> ~21 FLOP/B of fragment
// traffic (vs ~8 for the 16x16 tile), which is what the L0/L2 path can
// actually feed on MI455X. Requires M%64==0, N%32==0, K%32==0, non-batched.
// ---------------------------------------------------------------------------
__global__ __launch_bounds__(256)
void k_gemm_wmma_blk(const u16* __restrict__ A, int lda,
                     const u16* __restrict__ W, int ldw,
                     void* __restrict__ Cv, int ldc,
                     const float* __restrict__ bias,
                     int Mtiles /*M/64*/, int Ntiles /*N/32*/, int Ksteps,
                     float alpha, int outBf16)
{
  const int wid  = threadIdx.x >> 5;
  const int lane = threadIdx.x & 31;
  long tile = (long)blockIdx.x * 8 + wid;
  long tot  = (long)Mtiles * Ntiles;
  if (tile >= tot) return;
  const int mt = (int)(tile / Ntiles);
  const int nt = (int)(tile % Ntiles);

  const int row = lane & 15, hi = lane >> 4;
  const u16* Ap = A + (long)(mt * 64 + row) * lda + hi * 8;
  const u16* Wp = W + (long)(nt * 32 + row) * ldw + hi * 8;
  const long strideA16 = (long)16 * lda;   // next 16-row A fragment
  const long strideW16 = (long)16 * ldw;   // next 16-col B fragment

  f32x8 acc[4][2];
#pragma unroll
  for (int i = 0; i < 4; ++i)
#pragma unroll
    for (int j = 0; j < 2; ++j)
      acc[i][j] = (f32x8){0.f, 0.f, 0.f, 0.f, 0.f, 0.f, 0.f, 0.f};

  for (int kk = 0; kk < Ksteps; ++kk) {
    Frag a[4], b[2];
#pragma unroll
    for (int i = 0; i < 4; ++i) {
      const u16* p = Ap + i * strideA16;
      a[i].q[0] = *(const u32x4*)(p);
      a[i].q[1] = *(const u32x4*)(p + 16);
    }
#pragma unroll
    for (int j = 0; j < 2; ++j) {
      const u16* p = Wp + j * strideW16;
      b[j].q[0] = *(const u32x4*)(p);
      b[j].q[1] = *(const u32x4*)(p + 16);
    }
    __builtin_prefetch(Ap + 64, 0, 1);
    __builtin_prefetch(Wp + 64, 0, 1);
    Ap += 32; Wp += 32;
#pragma unroll
    for (int i = 0; i < 4; ++i)
#pragma unroll
      for (int j = 0; j < 2; ++j)
        acc[i][j] = __builtin_amdgcn_wmma_f32_16x16x32_bf16(
            false, a[i].v, false, b[j].v, (short)0, acc[i][j], false, false);
  }

#pragma unroll
  for (int j = 0; j < 2; ++j) {
    const int ccol = nt * 32 + j * 16 + row;
    const float bv = bias ? bias[ccol] : 0.f;
#pragma unroll
    for (int i = 0; i < 4; ++i) {
      if (outBf16) {
        u16* C = (u16*)Cv;
#pragma unroll
        for (int v = 0; v < 8; ++v) {
          int r = mt * 64 + i * 16 + v + hi * 8;
          C[(long)r * ldc + ccol] = f2bf(acc[i][j][v] * alpha + bv);
        }
      } else {
        float* C = (float*)Cv;
#pragma unroll
        for (int v = 0; v < 8; ++v) {
          int r = mt * 64 + i * 16 + v + hi * 8;
          C[(long)r * ldc + ccol] = acc[i][j][v] * alpha + bv;
        }
      }
    }
  }
}

// ---------------------------------------------------------------------------
// One LSTM time step, fully fused in ONE workgroup (1024 threads = 32 waves):
//   gates[32,2048] = h_state[32,512](bf16) @ Whh[2048,512]^T + xg[:,t,:]
// accumulated with WMMA into 256KB of dynamic LDS (CDNA5: 320KB/WGP),
// then the gate nonlinearity + state update behind one barrier.
// Wave w: row-tile (w&1), col-tiles (w>>1)*8 .. +8  ->  8 accumulators,
// A fragment reused 8x per K-step.
// ---------------------------------------------------------------------------
__global__ __launch_bounds__(1024)
void k_lstm_step(const u16* __restrict__ Whh, const float* __restrict__ xg,
                 int t, int T, u16* __restrict__ h_state,
                 float* __restrict__ c_state, u16* __restrict__ h_out)
{
  extern __shared__ float gates[];           // 32*2048 floats = 256 KB
  const int wid  = threadIdx.x >> 5;
  const int lane = threadIdx.x & 31;
  const int row = lane & 15, hi = lane >> 4;
  const int rtile = wid & 1;                 // rows rtile*16 .. +16
  const int cg    = wid >> 1;                // 16 groups x 8 n-tiles = 128

  const u16* Ap = h_state + (long)(rtile * 16 + row) * 512 + hi * 8;
  f32x8 acc[8];
#pragma unroll
  for (int i = 0; i < 8; ++i) acc[i] = (f32x8){0.f,0.f,0.f,0.f,0.f,0.f,0.f,0.f};

  for (int kk = 0; kk < 16; ++kk) {
    Frag a;
    a.q[0] = *(const u32x4*)(Ap + kk * 32);
    a.q[1] = *(const u32x4*)(Ap + kk * 32 + 16);
#pragma unroll
    for (int i = 0; i < 8; ++i) {
      const int n0 = (cg * 8 + i) * 16;
      const u16* Wp = Whh + (long)(n0 + row) * 512 + kk * 32 + hi * 8;
      Frag b;
      b.q[0] = *(const u32x4*)(Wp);
      b.q[1] = *(const u32x4*)(Wp + 16);
      acc[i] = __builtin_amdgcn_wmma_f32_16x16x32_bf16(false, a.v, false, b.v,
                                                       (short)0, acc[i], false, false);
    }
  }
  // scatter C tiles + xg into LDS gate matrix
#pragma unroll
  for (int i = 0; i < 8; ++i) {
    const int c = (cg * 8 + i) * 16 + row;
#pragma unroll
    for (int v = 0; v < 8; ++v) {
      const int b = rtile * 16 + v + hi * 8;
      gates[b * 2048 + c] = acc[i][v] + xg[((long)b * T + t) * 2048 + c];
    }
  }
  __syncthreads();

  // pointwise: i,f,g,o chunks of 512; c' = sig(f)*c + sig(i)*tanh(g); h = sig(o)*tanh(c')
#pragma unroll
  for (int it = 0; it < 16; ++it) {
    int idx = threadIdx.x + it * 1024;      // 32*512 states
    int b = idx >> 9, j = idx & 511;
    float ig = gates[b * 2048 + j];
    float fg = gates[b * 2048 + 512 + j];
    float gg = gates[b * 2048 + 1024 + j];
    float og = gates[b * 2048 + 1536 + j];
    float c = sigmoidf_(fg) * c_state[idx] + sigmoidf_(ig) * tanhf(gg);
    c_state[idx] = c;
    float h = sigmoidf_(og) * tanhf(c);
    u16 hb = f2bf(h);
    h_state[idx] = hb;
    h_out[((long)b * T + t) * 512 + j] = hb;
  }
}

// ---------------------------------------------------------------------------
// Softmax over S=256 per row, one wave per row, shuffle reductions. bf16 out.
// ---------------------------------------------------------------------------
__global__ __launch_bounds__(256)
void k_softmax256(const float* __restrict__ Sm, u16* __restrict__ P, long nrows)
{
  int wid = threadIdx.x >> 5, lane = threadIdx.x & 31;
  long r = (long)blockIdx.x * 8 + wid;
  if (r >= nrows) return;
  const float* s = Sm + r * 256;
  float v[8], m = -1e30f;
#pragma unroll
  for (int k = 0; k < 8; ++k) { v[k] = s[lane + k * 32]; m = fmaxf(m, v[k]); }
  for (int off = 16; off; off >>= 1) m = fmaxf(m, __shfl_xor(m, off, 32));
  float sum = 0.f;
#pragma unroll
  for (int k = 0; k < 8; ++k) { v[k] = __expf(v[k] - m); sum += v[k]; }
  for (int off = 16; off; off >>= 1) sum += __shfl_xor(sum, off, 32);
  float inv = 1.f / sum;
  u16* p = P + r * 256;
#pragma unroll
  for (int k = 0; k < 8; ++k) p[lane + k * 32] = f2bf(v[k] * inv);
}

// --------------------------- small utility kernels -------------------------
__global__ void k_f32_to_bf16(const float* __restrict__ s, u16* __restrict__ d, long n) {
  long i = (long)blockIdx.x * blockDim.x + threadIdx.x;
  if (i < n) d[i] = f2bf(s[i]);
}
// dst[N][Kp] = src[N][K] (zero-padded K)
__global__ void k_conv_pad(const float* __restrict__ src, u16* __restrict__ dst,
                           int K, int Kp, long n) {
  long i = (long)blockIdx.x * blockDim.x + threadIdx.x;
  if (i >= n) return;
  long nrow = i / Kp; int k = (int)(i % Kp);
  dst[i] = (k < K) ? f2bf(src[nrow * K + k]) : (u16)0;
}
// dst[N][Kp] = src[K][N]^T (zero-padded K)
__global__ void k_conv_trans(const float* __restrict__ src, u16* __restrict__ dst,
                             int K, int N, int Kp, long n) {
  long i = (long)blockIdx.x * blockDim.x + threadIdx.x;
  if (i >= n) return;
  long nrow = i / Kp; int k = (int)(i % Kp);
  dst[i] = (k < K) ? f2bf(src[(long)k * N + nrow]) : (u16)0;
}
// dec_in = shift(mel_targets) ; A0 = bf16 pad-96 copy
__global__ void k_build_dec_in(const float* __restrict__ mel_t, float* __restrict__ dec_in,
                               u16* __restrict__ A0, int T, long n) {
  long i = (long)blockIdx.x * blockDim.x + threadIdx.x;
  if (i >= n) return;                      // n = BT*96
  long r = i / 96; int k = (int)(i % 96);
  int t = (int)(r % T);
  float v = 0.f;
  if (k < 80 && t > 0) v = mel_t[(r - 1) * 80 + k];
  A0[i] = f2bf(v);
  if (k < 80) dec_in[r * 80 + k] = v;
}
// lstm_in[:,512:608] = [bf16(dec_in), zeros]
__global__ void k_lstmin_tail(const float* __restrict__ dec_in, u16* __restrict__ li, long n) {
  long i = (long)blockIdx.x * blockDim.x + threadIdx.x;
  if (i >= n) return;                      // n = BT*96
  long r = i / 96; int k = (int)(i % 96);
  float v = (k < 80) ? dec_in[r * 80 + k] : 0.f;
  li[r * 608 + 512 + k] = f2bf(v);
}
// Vt[b,h,d,s] <- vbuf[b,s,h*64+d]   (B-side for context GEMM)
__global__ void k_v_trans(const u16* __restrict__ vbuf, u16* __restrict__ Vt, long n) {
  long i = (long)blockIdx.x * blockDim.x + threadIdx.x;
  if (i >= n) return;                      // n = 32*8*64*256
  int s = (int)(i & 255);
  int d = (int)((i >> 8) & 63);
  int h = (int)((i >> 14) & 7);
  int b = (int)(i >> 17);
  Vt[i] = vbuf[((long)(b * 256 + s)) * 512 + h * 64 + d];
}
// conv weight repack: Wc[o][tau*C+c] = pw[o][c][tau], K padded
__global__ void k_conv_repack(const float* __restrict__ pw, u16* __restrict__ Wc,
                              int C, int Kp, long n) {
  long i = (long)blockIdx.x * blockDim.x + threadIdx.x;
  if (i >= n) return;
  long o = i / Kp; int k = (int)(i % Kp);
  u16 v = 0;
  if (k < 5 * C) { int tau = k / C, c = k % C; v = f2bf(pw[(o * C + c) * 5 + tau]); }
  Wc[i] = v;
}
// im2col for K=5 'SAME' conv, channel-last x[B,T,C] -> dst[BT, Kp]
__global__ void k_im2col(const u16* __restrict__ x, u16* __restrict__ dst,
                         int C, int Kp, int T, long n) {
  long i = (long)blockIdx.x * blockDim.x + threadIdx.x;
  if (i >= n) return;
  long r = i / Kp; int k = (int)(i % Kp);
  u16 v = 0;
  if (k < 5 * C) {
    int tau = k / C, c = k % C;
    int t = (int)(r % T) + tau - 2;
    if (t >= 0 && t < T) v = x[(r - (r % T) + t) * (long)C + c];
  }
  dst[i] = v;
}
__global__ void k_addvec(const float* a, const float* b, float* o, int n) {
  int i = blockIdx.x * blockDim.x + threadIdx.x;
  if (i < n) o[i] = a[i] + b[i];
}
__global__ void k_add(const float* __restrict__ a, const float* __restrict__ b,
                      float* __restrict__ o, long n) {
  long i = (long)blockIdx.x * blockDim.x + threadIdx.x;
  if (i < n) o[i] = a[i] + b[i];
}
__global__ void k_zero_u16(u16* p, long n) {
  long i = (long)blockIdx.x * blockDim.x + threadIdx.x;
  if (i < n) p[i] = 0;
}
__global__ void k_zero_f32(float* p, long n) {
  long i = (long)blockIdx.x * blockDim.x + threadIdx.x;
  if (i < n) p[i] = 0.f;
}
// BatchNorm training-mode stats per channel (channel-last y[rows,C])
__global__ __launch_bounds__(256)
void k_bn_stats(const float* __restrict__ y, float* __restrict__ mu,
                float* __restrict__ var, int C, long rows) {
  __shared__ float s1[256], s2[256];
  int c = blockIdx.x, tid = threadIdx.x;
  float a = 0.f, b = 0.f;
  for (long r = tid; r < rows; r += 256) {
    float v = y[r * C + c]; a += v; b += v * v;
  }
  s1[tid] = a; s2[tid] = b;
  __syncthreads();
  for (int off = 128; off > 0; off >>= 1) {
    if (tid < off) { s1[tid] += s1[tid + off]; s2[tid] += s2[tid + off]; }
    __syncthreads();
  }
  if (tid == 0) {
    float m = s1[0] / (float)rows;
    mu[c] = m;
    var[c] = s2[0] / (float)rows - m * m;
  }
}
__global__ void k_bn_tanh(const float* __restrict__ y, const float* __restrict__ mu,
                          const float* __restrict__ var, const float* __restrict__ g,
                          const float* __restrict__ be, u16* __restrict__ out,
                          int C, long n) {
  long i = (long)blockIdx.x * blockDim.x + threadIdx.x;
  if (i >= n) return;
  int c = (int)(i % C);
  float xn = (y[i] - mu[c]) * rsqrtf(var[c] + 1e-5f) * g[c] + be[c];
  out[i] = f2bf(tanhf(xn));
}
// stop projection: one wave per row, 512-dot + shuffle reduce
__global__ __launch_bounds__(256)
void k_stop(const u16* __restrict__ h2, const float* __restrict__ Ws,
            const float* __restrict__ bs, float* __restrict__ out, long rows) {
  int wid = threadIdx.x >> 5, lane = threadIdx.x & 31;
  long r = (long)blockIdx.x * 8 + wid;
  if (r >= rows) return;
  const u16* h = h2 + r * 512;
  float s = 0.f;
  for (int j = lane; j < 512; j += 32) s += bf2f(h[j]) * Ws[j];
  for (int off = 16; off; off >>= 1) s += __shfl_xor(s, off, 32);
  if (lane == 0) out[r] = s + bs[0];
}

// ---------------------------------------------------------------------------
extern "C" void kernel_launch(void* const* d_in, const int* in_sizes, int n_in,
                              void* d_out, int out_size, void* d_ws, size_t ws_size,
                              hipStream_t stream) {
  (void)in_sizes; (void)n_in; (void)out_size; (void)ws_size;
  const float* enc    = (const float*)d_in[0];
  const float* melT   = (const float*)d_in[1];
  const float* W_in   = (const float*)d_in[2];  const float* b_in  = (const float*)d_in[3];
  const float* Wq     = (const float*)d_in[4];  const float* bq    = (const float*)d_in[5];
  const float* Wk     = (const float*)d_in[6];  const float* bk    = (const float*)d_in[7];
  const float* Wv     = (const float*)d_in[8];  const float* bv    = (const float*)d_in[9];
  const float* Wo     = (const float*)d_in[10]; const float* bo    = (const float*)d_in[11];
  const float* W_ih0  = (const float*)d_in[12]; const float* W_hh0 = (const float*)d_in[13];
  const float* b_ih0  = (const float*)d_in[14]; const float* b_hh0 = (const float*)d_in[15];
  const float* W_ih1  = (const float*)d_in[16]; const float* W_hh1 = (const float*)d_in[17];
  const float* b_ih1  = (const float*)d_in[18]; const float* b_hh1 = (const float*)d_in[19];
  const float* W_mel  = (const float*)d_in[20]; const float* b_mel = (const float*)d_in[21];
  const float* W_stop = (const float*)d_in[22]; const float* b_stop= (const float*)d_in[23];
  const float* pw[5]  = {(const float*)d_in[24], (const float*)d_in[26], (const float*)d_in[28],
                         (const float*)d_in[30], (const float*)d_in[32]};
  const float* pb[5]  = {(const float*)d_in[25], (const float*)d_in[27], (const float*)d_in[29],
                         (const float*)d_in[31], (const float*)d_in[33]};
  const float* bg[4]  = {(const float*)d_in[34], (const float*)d_in[36],
                         (const float*)d_in[38], (const float*)d_in[40]};
  const float* bbe[4] = {(const float*)d_in[35], (const float*)d_in[37],
                         (const float*)d_in[39], (const float*)d_in[41]};
  float* outF = (float*)d_out;   // [mel 1024000][mel_postnet 1024000][stop 12800]

  // ----- workspace carve (requires ws_size >= ~380MB; big regions shared) ---
  char* base = (char*)d_ws; size_t cur = 0;
  auto alloc = [&](size_t bytes) -> void* {
    cur = (cur + 255) & ~(size_t)255;
    void* p = base + cur; cur += bytes; return p;
  };
  float* dec_in  = (float*)alloc(BT * 80 * 4);
  u16*   A0      = (u16*)  alloc(BT * 96 * 2);
  u16*   encbf   = (u16*)  alloc(BS * 512 * 2);
  u16*   WinT    = (u16*)  alloc(512 * 96 * 2);
  u16*   WqT     = (u16*)  alloc(512 * 512 * 2);
  u16*   WkT     = (u16*)  alloc(512 * 512 * 2);
  u16*   WvT     = (u16*)  alloc(512 * 512 * 2);
  u16*   WoT     = (u16*)  alloc(512 * 512 * 2);
  u16*   Wih0b   = (u16*)  alloc(2048L * 608 * 2);
  u16*   Whh0b   = (u16*)  alloc(2048L * 512 * 2);
  u16*   Wih1b   = (u16*)  alloc(2048L * 512 * 2);
  u16*   Whh1b   = (u16*)  alloc(2048L * 512 * 2);
  u16*   WmelT   = (u16*)  alloc(80 * 512 * 2);
  u16*   Wc0     = (u16*)  alloc(512L * 416 * 2);
  u16*   Wc1     = (u16*)  alloc(512L * 2560 * 2);
  u16*   Wc2     = (u16*)  alloc(512L * 2560 * 2);
  u16*   Wc3     = (u16*)  alloc(512L * 2560 * 2);
  u16*   Wc4     = (u16*)  alloc(80L * 2560 * 2);
  float* bias0   = (float*)alloc(2048 * 4);
  float* bias1   = (float*)alloc(2048 * 4);
  u16*   qin     = (u16*)  alloc(BT * 512 * 2);
  u16*   qbuf    = (u16*)  alloc(BT * 512 * 2);
  u16*   kbuf    = (u16*)  alloc(BS * 512 * 2);
  u16*   vbuf    = (u16*)  alloc(BS * 512 * 2);
  u16*   Vt      = (u16*)  alloc(BS * 512 * 2);
  float* big1    = (float*)alloc(BT * 2048 * 4);   // scores f32, later xg f32
  u16*   big2    = (u16*)  alloc(BT * 2560 * 2);   // attn bf16, later im2col
  u16*   ctxb    = (u16*)  alloc(BT * 512 * 2);
  u16*   lstm_in = (u16*)  alloc(BT * 608 * 2);
  u16*   h1buf   = (u16*)  alloc(BT * 512 * 2);
  u16*   h2buf   = (u16*)  alloc(BT * 512 * 2);
  u16*   h_state = (u16*)  alloc(32 * 512 * 2);
  float* c_state = (float*)alloc(32 * 512 * 4);
  u16*   xa      = (u16*)  alloc(BT * 512 * 2);
  u16*   xb      = (u16*)  alloc(BT * 512 * 2);
  float* convY   = (float*)alloc(BT * 512 * 4);
  float* bnmu    = (float*)alloc(512 * 4);
  float* bnvar   = (float*)alloc(512 * 4);

  auto gblk = [](long n) { return dim3((unsigned)((n + 255) / 256)); };
  // Routed GEMM: 64x32 register-blocked wave tile when shapes allow,
  // generic 16x16 tile (with batching) otherwise.
  auto gemm = [&](const u16* A, int lda, long sA0, long sA1,
                  const u16* W, int ldw, long sW0, long sW1,
                  void* C, int ldc, long sC0, long sC1,
                  const float* bias, int M, int N, int K,
                  int batch, int zdiv, float alpha, int outBf16) {
    if (batch == 1 && (M % 64) == 0 && (N % 32) == 0) {
      long tiles = (long)(M / 64) * (N / 32);
      dim3 grid((unsigned)((tiles + 7) / 8), 1);
      k_gemm_wmma_blk<<<grid, 256, 0, stream>>>(A, lda, W, ldw, C, ldc, bias,
                                                M / 64, N / 32, K / 32,
                                                alpha, outBf16);
    } else {
      long tiles = (long)(M / 16) * (N / 16);
      dim3 grid((unsigned)((tiles + 7) / 8), (unsigned)batch);
      k_gemm_wmma<<<grid, 256, 0, stream>>>(A, lda, sA0, sA1, W, ldw, sW0, sW1,
                                            C, ldc, sC0, sC1, bias,
                                            M / 16, N / 16, K / 32, zdiv, alpha, outBf16);
    }
  };

  // ----- input prep & weight conversion (every call; ws is poisoned) -------
  k_build_dec_in<<<gblk(BT * 96), 256, 0, stream>>>(melT, dec_in, A0, Tc, BT * 96);
  k_f32_to_bf16<<<gblk(BS * 512), 256, 0, stream>>>(enc, encbf, BS * 512);
  k_conv_trans<<<gblk(512L * 96), 256, 0, stream>>>(W_in, WinT, 80, 512, 96, 512L * 96);
  k_conv_trans<<<gblk(512L * 512), 256, 0, stream>>>(Wq, WqT, 512, 512, 512, 512L * 512);
  k_conv_trans<<<gblk(512L * 512), 256, 0, stream>>>(Wk, WkT, 512, 512, 512, 512L * 512);
  k_conv_trans<<<gblk(512L * 512), 256, 0, stream>>>(Wv, WvT, 512, 512, 512, 512L * 512);
  k_conv_trans<<<gblk(512L * 512), 256, 0, stream>>>(Wo, WoT, 512, 512, 512, 512L * 512);
  k_conv_pad<<<gblk(2048L * 608), 256, 0, stream>>>(W_ih0, Wih0b, 592, 608, 2048L * 608);
  k_conv_pad<<<gblk(2048L * 512), 256, 0, stream>>>(W_hh0, Whh0b, 512, 512, 2048L * 512);
  k_conv_pad<<<gblk(2048L * 512), 256, 0, stream>>>(W_ih1, Wih1b, 512, 512, 2048L * 512);
  k_conv_pad<<<gblk(2048L * 512), 256, 0, stream>>>(W_hh1, Whh1b, 512, 512, 2048L * 512);
  k_conv_trans<<<gblk(80L * 512), 256, 0, stream>>>(W_mel, WmelT, 512, 80, 512, 80L * 512);
  k_conv_repack<<<gblk(512L * 416), 256, 0, stream>>>(pw[0], Wc0, 80, 416, 512L * 416);
  k_conv_repack<<<gblk(512L * 2560), 256, 0, stream>>>(pw[1], Wc1, 512, 2560, 512L * 2560);
  k_conv_repack<<<gblk(512L * 2560), 256, 0, stream>>>(pw[2], Wc2, 512, 2560, 512L * 2560);
  k_conv_repack<<<gblk(512L * 2560), 256, 0, stream>>>(pw[3], Wc3, 512, 2560, 512L * 2560);
  k_conv_repack<<<gblk(80L * 2560), 256, 0, stream>>>(pw[4], Wc4, 512, 2560, 80L * 2560);
  k_addvec<<<8, 256, 0, stream>>>(b_ih0, b_hh0, bias0, 2048);
  k_addvec<<<8, 256, 0, stream>>>(b_ih1, b_hh1, bias1, 2048);

  // ----- attention front-end (all bf16 WMMA GEMMs) -------------------------
  gemm(A0, 96, 0, 0, WinT, 96, 0, 0, qin, 512, 0, 0, b_in, (int)BT, 512, 96, 1, 1, 1.f, 1);
  gemm(qin, 512, 0, 0, WqT, 512, 0, 0, qbuf, 512, 0, 0, bq, (int)BT, 512, 512, 1, 1, 1.f, 1);
  gemm(encbf, 512, 0, 0, WkT, 512, 0, 0, kbuf, 512, 0, 0, bk, (int)BS, 512, 512, 1, 1, 1.f, 1);
  gemm(encbf, 512, 0, 0, WvT, 512, 0, 0, vbuf, 512, 0, 0, bv, (int)BS, 512, 512, 1, 1, 1.f, 1);
  k_v_trans<<<gblk(BS * 512), 256, 0, stream>>>(vbuf, Vt, BS * 512);

  // scores[b,h,T,S] = (Qh Kh^T)/8 : batched over z=b*8+h, K-side is kbuf directly
  gemm(qbuf, 512, (long)Tc * 512, 64,
       kbuf, 512, (long)Sc * 512, 64,
       big1, 256, 8L * Tc * 256, (long)Tc * 256,
       nullptr, Tc, Sc, 64, Bc * Hc, Hc, 0.125f, 0);
  u16* attn = big2;
  k_softmax256<<<gblk((long)Bc * Hc * Tc * 8) /* 8 rows/blk */, 256, 0, stream>>>(
      big1, attn, (long)Bc * Hc * Tc);
  // ctx[b,t,h*64+d] = attn @ V : B-side is Vt[b,h,64,256]
  gemm(attn, 256, 8L * Tc * 256, (long)Tc * 256,
       Vt, 256, 8L * 64 * 256, 64L * 256,
       ctxb, 512, (long)Tc * 512, 64,
       nullptr, Tc, 64, 256, Bc * Hc, Hc, 1.f, 1);
  // attn_out -> lstm_in[:, 0:512]; tail = [dec_in, zero-pad]
  gemm(ctxb, 512, 0, 0, WoT, 512, 0, 0, lstm_in, 608, 0, 0, bo, (int)BT, 512, 512, 1, 1, 1.f, 1);
  k_lstmin_tail<<<gblk(BT * 96), 256, 0, stream>>>(dec_in, lstm_in, BT * 96);

  // ----- LSTM layer 0 ------------------------------------------------------
  float* xg = big1;   // reuse scores region
  gemm(lstm_in, 608, 0, 0, Wih0b, 608, 0, 0, xg, 2048, 0, 0, bias0,
       (int)BT, 2048, 608, 1, 1, 1.f, 0);
  k_zero_u16<<<gblk(32 * 512), 256, 0, stream>>>(h_state, 32 * 512);
  k_zero_f32<<<gblk(32 * 512), 256, 0, stream>>>(c_state, 32 * 512);
  for (int t = 0; t < Tc; ++t)
    k_lstm_step<<<1, 1024, 32 * 2048 * sizeof(float), stream>>>(
        Whh0b, xg, t, Tc, h_state, c_state, h1buf);

  // ----- LSTM layer 1 ------------------------------------------------------
  gemm(h1buf, 512, 0, 0, Wih1b, 512, 0, 0, xg, 2048, 0, 0, bias1,
       (int)BT, 2048, 512, 1, 1, 1.f, 0);
  k_zero_u16<<<gblk(32 * 512), 256, 0, stream>>>(h_state, 32 * 512);
  k_zero_f32<<<gblk(32 * 512), 256, 0, stream>>>(c_state, 32 * 512);
  for (int t = 0; t < Tc; ++t)
    k_lstm_step<<<1, 1024, 32 * 2048 * sizeof(float), stream>>>(
        Whh1b, xg, t, Tc, h_state, c_state, h2buf);

  // ----- output projections ------------------------------------------------
  gemm(h2buf, 512, 0, 0, WmelT, 512, 0, 0, outF, 80, 0, 0, b_mel,
       (int)BT, 80, 512, 1, 1, 1.f, 0);                      // mel_outputs
  k_stop<<<gblk(BT * 8), 256, 0, stream>>>(h2buf, W_stop, b_stop, outF + 2048000, BT);

  // ----- postnet: conv-as-GEMM (im2col K=5) + training-mode BN + tanh ------
  u16* Aconv = big2;  // attn region dead now
  k_f32_to_bf16<<<gblk(BT * 80), 256, 0, stream>>>(outF, xa, BT * 80);
  // L0: 80 -> 512
  k_im2col<<<gblk(BT * 416), 256, 0, stream>>>(xa, Aconv, 80, 416, Tc, BT * 416);
  gemm(Aconv, 416, 0, 0, Wc0, 416, 0, 0, convY, 512, 0, 0, pb[0], (int)BT, 512, 416, 1, 1, 1.f, 0);
  k_bn_stats<<<512, 256, 0, stream>>>(convY, bnmu, bnvar, 512, BT);
  k_bn_tanh<<<gblk(BT * 512), 256, 0, stream>>>(convY, bnmu, bnvar, bg[0], bbe[0], xb, 512, BT * 512);
  // L1..L3: 512 -> 512 (ping-pong xa/xb)
  u16* cin = xb; u16* cout = xa;
  u16* Wcl[3] = {Wc1, Wc2, Wc3};
  for (int l = 0; l < 3; ++l) {
    k_im2col<<<gblk(BT * 2560), 256, 0, stream>>>(cin, Aconv, 512, 2560, Tc, BT * 2560);
    gemm(Aconv, 2560, 0, 0, Wcl[l], 2560, 0, 0, convY, 512, 0, 0, pb[l + 1],
         (int)BT, 512, 2560, 1, 1, 1.f, 0);
    k_bn_stats<<<512, 256, 0, stream>>>(convY, bnmu, bnvar, 512, BT);
    k_bn_tanh<<<gblk(BT * 512), 256, 0, stream>>>(convY, bnmu, bnvar, bg[l + 1], bbe[l + 1],
                                                  cout, 512, BT * 512);
    u16* tmp = cin; cin = cout; cout = tmp;
  }
  // L4: 512 -> 80, no BN/tanh; residual add -> mel_postnet
  k_im2col<<<gblk(BT * 2560), 256, 0, stream>>>(cin, Aconv, 512, 2560, Tc, BT * 2560);
  gemm(Aconv, 2560, 0, 0, Wc4, 2560, 0, 0, convY, 80, 0, 0, pb[4], (int)BT, 80, 2560, 1, 1, 1.f, 0);
  k_add<<<gblk(BT * 80), 256, 0, stream>>>(outF, convY, outF + 1024000, BT * 80);
}
